// AuxiliaryModel_83906481095166
// MI455X (gfx1250) — matrix operand order repaired and checked by reading, not compiled, not run on hardware
//
#include <hip/hip_runtime.h>

// MI455X (gfx1250) fused R-BERT heads:
//   binary = tanh(x) @ W_bin + b_bin ; route = argmax ; logits = route ? x@W1+b1 : x@W0+b0
// Roofline: 256MB x-read / 23.3 TB/s ~= 11.3us floor. fp32 WMMA (K=4) can't keep up;
// use bf16 WMMA (K=32) with 3-term Dekker split (hi*hi + lo*hi + hi*lo, fp32 accum)
// for ~fp32 accuracy at ~2.2 PFLOPS-bf16 demand (well under dense bf16 ceiling).
// Split weights staged transposed in LDS (264KB of the 320KB/WGP).

typedef __bf16 bf16_t;
typedef __attribute__((ext_vector_type(16))) __bf16 v16bf;
typedef __attribute__((ext_vector_type(8)))  __bf16 bf16x8;
typedef __attribute__((ext_vector_type(8)))  float  v8f;
typedef __attribute__((ext_vector_type(4)))  float  f32x4;

static constexpr int kH     = 1024;  // hidden dim (K)
static constexpr int kCols  = 66;    // 2 bin | 30 W0 | 30 W1 | 4 zero pad
static constexpr int kBlkT  = 256;   // 8 waves
static constexpr int kRowsW = 32;    // rows per wave (2 WMMA M-tiles)
static constexpr int kRowsB = 256;   // rows per block

__device__ __forceinline__ float fast_tanh(float x) {
  // tanh(x) = 1 - 2/(exp(2x)+1); v_exp_f32 + v_rcp_f32, inf/NaN-safe at extremes
  float e = __builtin_amdgcn_exp2f(x * 2.8853900817779268f); // 2*log2(e)
  return 1.0f - 2.0f * __builtin_amdgcn_rcpf(e + 1.0f);
}
__device__ __forceinline__ float bperm_f(int srcLane, float v) {
  return __int_as_float(__builtin_amdgcn_ds_bpermute(srcLane << 2, __float_as_int(v)));
}

__global__ __launch_bounds__(kBlkT, 1)
void rbert_heads_kernel(const float* __restrict__ X,
                        const float* __restrict__ Wbin, const float* __restrict__ Bbin,
                        const float* __restrict__ W0,   const float* __restrict__ B0,
                        const float* __restrict__ W1,   const float* __restrict__ B1,
                        float* __restrict__ outBin, float* __restrict__ outLog)
{
  extern __shared__ bf16_t smem[];
  bf16_t* wt_hi = smem;               // [kCols][kH], transposed weights, hi part
  bf16_t* wt_lo = smem + kCols * kH;  // lo part (Dekker split residual)

  const int tid = threadIdx.x;

  // ---- stage split weights into LDS (cols: 0-1 bin, 2-31 W0, 32-61 W1, 62-65 zero)
  for (int idx = tid; idx < kCols * kH; idx += kBlkT) {
    int nc = idx >> 10, k = idx & (kH - 1);
    float w = 0.0f;
    if (nc < 2)       w = Wbin[k * 2 + nc];
    else if (nc < 32) w = W0[k * 30 + (nc - 2)];
    else if (nc < 62) w = W1[k * 30 + (nc - 32)];
    bf16_t hi = (bf16_t)w;
    bf16_t lo = (bf16_t)(w - (float)hi);
    wt_hi[idx] = hi;
    wt_lo[idx] = lo;
  }
  __syncthreads();

  const int lane   = tid & 31;
  const int wave   = tid >> 5;
  const int n      = lane & 15;
  const int hiHalf = lane >> 4;            // wave32 half select
  const int koffA  = hiHalf * 8;           // A layout: lanes 0-15 K0..7/K16..23, 16-31 K8..15/K24..31
  const int koffB  = hiHalf * 8;           // B layout: lanes 0-15 K0..7/K16..23, 16-31 K8..15/K24..31
  const int rowBase = blockIdx.x * kRowsB + wave * kRowsW;

  const v8f vzero = {0.f,0.f,0.f,0.f,0.f,0.f,0.f,0.f};
  v8f acc[2][5];
#pragma unroll
  for (int mt = 0; mt < 2; ++mt)
#pragma unroll
    for (int t = 0; t < 5; ++t) acc[mt][t] = vzero;

  const int colBaseT[5] = {0, 2, 18, 34, 50};
  const int col0 = (n < 2) ? n : 62;       // tanh tile: pad lanes read zeroed column

#pragma unroll 1
  for (int kt = 0; kt < kH / 32; ++kt) {
    const int k0 = kt * 32;
    v16bf xh[2], xl[2], th[2], tl[2];
#pragma unroll
    for (int mt = 0; mt < 2; ++mt) {
      const float* xp = X + (size_t)(rowBase + mt * 16 + n) * kH + k0 + koffA;
      f32x4 a = __builtin_nontemporal_load((const f32x4*)(xp +  0));
      f32x4 b = __builtin_nontemporal_load((const f32x4*)(xp +  4));
      f32x4 c = __builtin_nontemporal_load((const f32x4*)(xp + 16));
      f32x4 d = __builtin_nontemporal_load((const f32x4*)(xp + 20));
      float v[16] = {a[0],a[1],a[2],a[3], b[0],b[1],b[2],b[3],
                     c[0],c[1],c[2],c[3], d[0],d[1],d[2],d[3]};
#pragma unroll
      for (int i = 0; i < 16; ++i) {
        float f   = v[i];
        bf16_t fh = (bf16_t)f;
        xh[mt][i] = fh;
        xl[mt][i] = (bf16_t)(f - (float)fh);
        float tt  = fast_tanh(f);
        bf16_t thh = (bf16_t)tt;
        th[mt][i] = thh;
        tl[mt][i] = (bf16_t)(tt - (float)thh);
      }
    }
#pragma unroll
    for (int t = 0; t < 5; ++t) {
      const int col = (t == 0) ? col0 : (colBaseT[t] + n);
      const bf16_t* ph = wt_hi + col * kH + k0 + koffB;
      const bf16_t* pl = wt_lo + col * kH + k0 + koffB;
      bf16x8 h0 = *(const bf16x8*)(ph);
      bf16x8 h1 = *(const bf16x8*)(ph + 16);
      bf16x8 l0 = *(const bf16x8*)(pl);
      bf16x8 l1 = *(const bf16x8*)(pl + 16);
      v16bf Bh = __builtin_shufflevector(h0, h1, 0,1,2,3,4,5,6,7,8,9,10,11,12,13,14,15);
      v16bf Bl = __builtin_shufflevector(l0, l1, 0,1,2,3,4,5,6,7,8,9,10,11,12,13,14,15);
#pragma unroll
      for (int mt = 0; mt < 2; ++mt) {
        v16bf Ah = (t == 0) ? th[mt] : xh[mt];
        v16bf Al = (t == 0) ? tl[mt] : xl[mt];
        acc[mt][t] = __builtin_amdgcn_wmma_f32_16x16x32_bf16(false, Ah, false, Bh, (short)0, acc[mt][t], false, false);
        acc[mt][t] = __builtin_amdgcn_wmma_f32_16x16x32_bf16(false, Al, false, Bh, (short)0, acc[mt][t], false, false);
        acc[mt][t] = __builtin_amdgcn_wmma_f32_16x16x32_bf16(false, Ah, false, Bl, (short)0, acc[mt][t], false, false);
      }
    }
  }

  // ---- epilogue: bias, binary output, argmax route via bpermute, routed logits
  const float bb = (n < 2) ? Bbin[n] : 0.0f;
  int jcol[4], isE[4], vc[4]; float bias[4];
#pragma unroll
  for (int t = 1; t < 5; ++t) {
    int c = colBaseT[t] + n;
    int e = (c >= 32);
    int j = e ? (c - 32) : (c - 2);
    jcol[t-1] = j; isE[t-1] = e; vc[t-1] = (c <= 61);
    int jc = (j > 29) ? 29 : j;                 // clamp for safe pad-lane bias load
    bias[t-1] = e ? B1[jc] : B0[jc];
  }

#pragma unroll
  for (int mt = 0; mt < 2; ++mt) {
    const int rbase = rowBase + mt * 16 + hiHalf * 8;   // C layout: vgpr r -> M = r (+8)
#pragma unroll
    for (int r = 0; r < 8; ++r) {
      float vb = acc[mt][0][r] + bb;
      if (n < 2) outBin[(size_t)(rbase + r) * 2 + n] = vb;
      int src = lane & 16;                      // lanes holding cols 0/1 of this half
      float v0 = bperm_f(src + 0, vb);
      float v1 = bperm_f(src + 1, vb);
      int route1 = (v1 > v0);                   // jnp.argmax: first max wins ties
#pragma unroll
      for (int t = 1; t < 5; ++t) {
        if (vc[t-1] && (isE[t-1] == route1)) {
          outLog[(size_t)(rbase + r) * 30 + jcol[t-1]] = acc[mt][t][r] + bias[t-1];
        }
      }
    }
  }
}

extern "C" void kernel_launch(void* const* d_in, const int* in_sizes, int n_in,
                              void* d_out, int out_size, void* d_ws, size_t ws_size,
                              hipStream_t stream) {
  const float* X    = (const float*)d_in[0];
  const float* Wbin = (const float*)d_in[1];
  const float* Bbin = (const float*)d_in[2];
  const float* W0   = (const float*)d_in[3];
  const float* B0   = (const float*)d_in[4];
  const float* W1   = (const float*)d_in[5];
  const float* B1   = (const float*)d_in[6];
  float* out = (float*)d_out;

  const int Brows = in_sizes[0] / kH;               // 65536
  float* outBin = out;
  float* outLog = out + (size_t)Brows * 2;

  dim3 grid(Brows / kRowsB), block(kBlkT);
  size_t ldsBytes = (size_t)2 * kCols * kH * sizeof(bf16_t);  // 264 KB < 320 KB/WGP
  rbert_heads_kernel<<<grid, block, ldsBytes, stream>>>(X, Wbin, Bbin, W0, B0, W1, B1,
                                                        outBin, outLog);
}
